// EbmPosterior_68358699483926
// MI455X (gfx1250) — matrix operand order repaired and checked by reading, not compiled, not run on hardware
//
#include <hip/hip_runtime.h>
#include <hip/hip_bf16.h>
#include <math.h>

// Problem constants from the reference
#define BB 512
#define MM 8
#define SS 4
#define KK 64
#define TH 16
#define ZZ 16
#define HH 16
#define LOG2PI 1.8378770664093453f

typedef __attribute__((ext_vector_type(16))) _Float16 v16h;
typedef __attribute__((ext_vector_type(8)))  float    v8f;

// ---------------------------------------------------------------------------
// Phase 1: per-b pi-head MLP -> mu, std, sum(log std); plus t_part[b,m,h]
// Tiny: 512 threads, ~3K FLOP each. Pure VALU.
// ---------------------------------------------------------------------------
__global__ __launch_bounds__(256) void ebm_phase1(
    const float* __restrict__ thetas,     // (B,M,TH)
    const float* __restrict__ pi1_w, const float* __restrict__ pi1_b,
    const float* __restrict__ pimu_w, const float* __restrict__ pimu_b,
    const float* __restrict__ piprec_w, const float* __restrict__ piprec_b,
    const float* __restrict__ e1_w,       // (H, TH+Z) row-major, W_t = cols 0..15
    float* __restrict__ mu_out,           // (B,Z)
    float* __restrict__ std_out,          // (B,Z)
    float* __restrict__ sls_out,          // (B)  sum_z log(std)
    float* __restrict__ tpart_out)        // (B,M,H)
{
    int b = blockIdx.x * 256 + threadIdx.x;
    if (b >= BB) return;
    const float* th = thetas + (size_t)b * MM * TH;

    float tsum[TH];
#pragma unroll
    for (int t = 0; t < TH; ++t) {
        float a = 0.f;
#pragma unroll
        for (int m = 0; m < MM; ++m) a += th[m * TH + t];
        tsum[t] = a;
    }
    float h[HH];
#pragma unroll
    for (int j = 0; j < HH; ++j) {
        float a = pi1_b[j];
#pragma unroll
        for (int t = 0; t < TH; ++t) a += tsum[t] * pi1_w[j * TH + t];
        h[j] = fmaxf(a, 0.f);
    }
    float sls = 0.f;
#pragma unroll
    for (int z = 0; z < ZZ; ++z) {
        float m0 = pimu_b[z], pr = piprec_b[z];
#pragma unroll
        for (int j = 0; j < HH; ++j) {
            m0 += h[j] * pimu_w[z * HH + j];
            pr += h[j] * piprec_w[z * HH + j];
        }
        float prec = __expf(pr) + 0.01f;
        float var  = 1.f / (1.f + prec);
        float muv  = var * prec * m0;
        float st   = sqrtf(var);
        mu_out[b * ZZ + z]  = muv;
        std_out[b * ZZ + z] = st;
        sls += __logf(st);
    }
    sls_out[b] = sls;

    // t_part[b,m,h] = sum_t thetas[b,m,t] * e1_w[h, t]
#pragma unroll
    for (int m = 0; m < MM; ++m)
#pragma unroll
        for (int hh = 0; hh < HH; ++hh) {
            float a = 0.f;
#pragma unroll
            for (int t = 0; t < TH; ++t) a += th[m * TH + t] * e1_w[hh * (TH + ZZ) + t];
            tpart_out[((size_t)b * MM + m) * HH + hh] = a;
        }
}

// ---------------------------------------------------------------------------
// Phase 2: hot kernel. One wave32 handles 16 items (fixed b,s; k=k0..k0+15).
// WMMA#1: z_part(h x item) = W_z(h x z) * samples(z x item)     [16x16x32 f16]
// Per m:  WMMA#2: h2(g x item) = e2_w(g x h) * relu(h1)(h x item)
// Contraction dims padded 16->32 with zeros (upper halves / lanes 16-31 = 0).
// ---------------------------------------------------------------------------
__global__ __launch_bounds__(256) void ebm_phase2(
    const float* __restrict__ eps,        // (S,K,B,Z)
    const unsigned char* __restrict__ nan_mask, // (B,M) bools
    const float* __restrict__ e1_w,       // (H, TH+Z); W_z = cols 16..31
    const float* __restrict__ e1_b,
    const float* __restrict__ e2_w,       // (H,H)
    const float* __restrict__ e2_b,
    const float* __restrict__ e3_w,       // (1,H) flat 16
    const float* __restrict__ e3_b,       // (1)
    const float* __restrict__ mu,         // (B,Z)
    const float* __restrict__ stdv,       // (B,Z)
    const float* __restrict__ sls,        // (B)
    const float* __restrict__ t_part,     // (B,M,H)
    float* __restrict__ esum_out,         // (B,S,K)
    float* __restrict__ logiw_out)        // (B,S,K)
{
    const int tid  = threadIdx.x;
    const int lane = tid & 31;
    const int wave = tid >> 5;
    const int widx = blockIdx.x * 8 + wave;      // 8192 waves total
    const int b  = widx >> 4;
    const int s  = (widx >> 2) & 3;
    const int kg = widx & 3;
    const int k0 = kg * 16;
    const int lo = lane & 15;
    const bool hiHalf = (lane >= 16);

    // ---- constant A operands (loaded once, L2-resident) ----
    // A1 = W_z (M=h rows, K=z). Lanes 0-15: row h=lo, halves 0..15 = z 0..15;
    // lanes 16-31 (z 16..31) = 0.
    v16h aWz = {};
    {
        const float* wrow = e1_w + lo * (TH + ZZ) + TH;
#pragma unroll
        for (int j = 0; j < 16; ++j) aWz[j] = hiHalf ? (_Float16)0.f : (_Float16)wrow[j];
    }
    // A2 = e2_w (M=g rows, K=h), same padding scheme.
    v16h aE2 = {};
    {
        const float* wrow = e2_w + lo * HH;
#pragma unroll
        for (int j = 0; j < 16; ++j) aE2[j] = hiHalf ? (_Float16)0.f : (_Float16)wrow[j];
    }

    // ---- samples for this wave's 16 items (lane L and L+16 duplicate item L) ----
    const int item_k = k0 + lo;
    const float* epsPtr = eps + (((size_t)(s * KK + item_k) * BB + b) * ZZ);
    float smp[ZZ];
    float eps2sum = 0.f, s2sum = 0.f;
#pragma unroll
    for (int z = 0; z < ZZ; ++z) {
        float e  = epsPtr[z];
        float sm = mu[b * ZZ + z] + stdv[b * ZZ + z] * e;
        smp[z] = sm;
        eps2sum += e * e;
        s2sum   += sm * sm;
    }

    // B1 = samples (K=z rows, N=item cols): lane<16 col=lo, halves j = z=j.
    // Lanes 16-31 are z=16..31 -> zero.
    v16h b1;
#pragma unroll
    for (int j = 0; j < ZZ; ++j) b1[j] = hiHalf ? (_Float16)0.f : (_Float16)smp[j];

    v8f zc = {};
    zc = __builtin_amdgcn_wmma_f32_16x16x32_f16(false, aWz, false, b1,
                                                (short)0, zc, false, false);
    // D layout: zc[r] lanes0-15 -> z_part[h=r][item=lane]; lanes16-31 -> h=8+r.
    float zp[8], zu[8];
#pragma unroll
    for (int r = 0; r < 8; ++r) zp[r] = zc[r] + e1_b[hiHalf ? 8 + r : r];
#pragma unroll
    for (int r = 0; r < 8; ++r) zu[r] = __shfl_xor(zp[r], 16, 32);
    // lanes 0-15 now hold: zp[r] = z_part[h=r][item], zu[r] = z_part[h=8+r][item]

    float energyAcc = 0.f;
#pragma unroll
    for (int m = 0; m < MM; ++m) {
        const float* tp = t_part + ((size_t)b * MM + m) * HH;
        // B2 = relu(t_part + z_part) as (K=h rows, N=item cols); lanes>=16 zero.
        v16h b2;
#pragma unroll
        for (int j = 0; j < 8; ++j) {
            float v = fmaxf(zp[j] + tp[j], 0.f);
            b2[j] = hiHalf ? (_Float16)0.f : (_Float16)v;
        }
#pragma unroll
        for (int j = 0; j < 8; ++j) {
            float v = fmaxf(zu[j] + tp[8 + j], 0.f);
            b2[8 + j] = hiHalf ? (_Float16)0.f : (_Float16)v;
        }
        v8f d = {};
        d = __builtin_amdgcn_wmma_f32_16x16x32_f16(false, aE2, false, b2,
                                                   (short)0, d, false, false);
        // d[r]: lanes0-15 -> h2[g=r][item]; lanes16-31 -> h2[g=8+r][item]
        float p = 0.f;
#pragma unroll
        for (int r = 0; r < 8; ++r) {
            int g = hiHalf ? 8 + r : r;
            float h2v = fmaxf(d[r] + e2_b[g], 0.f);
            p += h2v * e3_w[g];
        }
        p += __shfl_xor(p, 16, 32);            // full 16-dot over g
        float energy = p + e3_b[0];
        float mval = nan_mask[b * MM + m] ? 0.f : 1.f;
        energyAcc += energy * mval;
    }

    // per-item scalars (lanes 0-15 authoritative; 16-31 are duplicates)
    float prior = -0.5f * s2sum - 8.f * LOG2PI;            // Z/2 * LOG2PI, Z=16
    float pilp  = -0.5f * eps2sum - sls[b] - 8.f * LOG2PI;
    float liw   = prior - energyAcc - pilp;
    if (!hiHalf) {
        int idx = (b * SS + s) * KK + item_k;
        esum_out[idx]  = energyAcc;
        logiw_out[idx] = liw;
    }
}

// ---------------------------------------------------------------------------
// Phase 3: one wave per (b,s). Gumbel argmax over K=64 (2 k per lane),
// logsumexp of log_iw, gather the winning sample, emit outputs.
// ---------------------------------------------------------------------------
__global__ __launch_bounds__(256) void ebm_phase3(
    const float* __restrict__ gumbel,     // (B,S,K)
    const float* __restrict__ eps,        // (S,K,B,Z)
    const float* __restrict__ mu,         // (B,Z)
    const float* __restrict__ stdv,       // (B,Z)
    const float* __restrict__ esum,       // (B,S,K)
    const float* __restrict__ logiw,      // (B,S,K)
    float* __restrict__ z_samples,        // (B,S,Z)
    float* __restrict__ log_probs)        // (B,S)
{
    const int tid  = threadIdx.x;
    const int lane = tid & 31;
    const int wave = tid >> 5;
    const int widx = blockIdx.x * 8 + wave;   // = b*S + s, 2048 total
    const int b = widx >> 2;
    const int s = widx & 3;

    const float* lw = logiw  + (size_t)widx * KK;
    const float* gu = gumbel + (size_t)widx * KK;

    float l0 = lw[lane], l1 = lw[lane + 32];
    float g0 = -__logf(-__logf(gu[lane]      + 1e-20f) + 1e-20f);
    float g1 = -__logf(-__logf(gu[lane + 32] + 1e-20f) + 1e-20f);
    float t0 = l0 + g0, t1 = l1 + g1;

    float bestv; int bestk;
    if (t0 >= t1) { bestv = t0; bestk = lane; }        // first-index tie-break
    else          { bestv = t1; bestk = lane + 32; }
#pragma unroll
    for (int off = 16; off >= 1; off >>= 1) {
        float ov = __shfl_xor(bestv, off, 32);
        int   ok = __shfl_xor(bestk, off, 32);
        if (ov > bestv || (ov == bestv && ok < bestk)) { bestv = ov; bestk = ok; }
    }

    float mx = fmaxf(l0, l1);
#pragma unroll
    for (int off = 16; off >= 1; off >>= 1) mx = fmaxf(mx, __shfl_xor(mx, off, 32));
    float se = __expf(l0 - mx) + __expf(l1 - mx);
#pragma unroll
    for (int off = 16; off >= 1; off >>= 1) se += __shfl_xor(se, off, 32);
    float log_avg_w = mx + __logf(se) - __logf(64.f);

    if (lane < ZZ) {
        float e = eps[(((size_t)(s * KK + bestk) * BB + b) * ZZ) + lane];
        z_samples[(size_t)widx * ZZ + lane] = mu[b * ZZ + lane] + stdv[b * ZZ + lane] * e;
    }
    if (lane == 0)
        log_probs[widx] = -esum[(size_t)widx * KK + bestk] - log_avg_w;
}

// ---------------------------------------------------------------------------
extern "C" void kernel_launch(void* const* d_in, const int* in_sizes, int n_in,
                              void* d_out, int out_size, void* d_ws, size_t ws_size,
                              hipStream_t stream) {
    const float* thetas   = (const float*)d_in[0];
    const unsigned char* nan_mask = (const unsigned char*)d_in[1]; // jax bool -> bytes
    const float* eps      = (const float*)d_in[2];
    const float* gumbel_u = (const float*)d_in[3];
    /* n_samples = d_in[4] (scalar, S=4 fixed) */
    const float* pi1_w    = (const float*)d_in[5];
    const float* pi1_b    = (const float*)d_in[6];
    const float* pimu_w   = (const float*)d_in[7];
    const float* pimu_b   = (const float*)d_in[8];
    const float* piprec_w = (const float*)d_in[9];
    const float* piprec_b = (const float*)d_in[10];
    const float* e1_w     = (const float*)d_in[11];
    const float* e1_b     = (const float*)d_in[12];
    const float* e2_w     = (const float*)d_in[13];
    const float* e2_b     = (const float*)d_in[14];
    const float* e3_w     = (const float*)d_in[15];
    const float* e3_b     = (const float*)d_in[16];

    // workspace layout (all f32, ~1.35 MB -> L2-resident on MI455X)
    char* ws = (char*)d_ws;
    float* mu_ws   = (float*)(ws);                                  // B*Z
    float* std_ws  = (float*)(ws + 32768);                          // B*Z
    float* sls_ws  = (float*)(ws + 65536);                          // B
    float* tp_ws   = (float*)(ws + 67584);                          // B*M*H
    float* es_ws   = (float*)(ws + 67584 + 262144);                 // B*S*K
    float* liw_ws  = (float*)(ws + 67584 + 262144 + 524288);        // B*S*K

    float* z_samples = (float*)d_out;                 // (B,S,Z) = 32768 floats
    float* log_probs = (float*)d_out + BB * SS * ZZ;  // (B,S)   = 2048 floats

    ebm_phase1<<<BB / 256, 256, 0, stream>>>(
        thetas, pi1_w, pi1_b, pimu_w, pimu_b, piprec_w, piprec_b, e1_w,
        mu_ws, std_ws, sls_ws, tp_ws);

    // 8192 waves (16 items each) = 1024 blocks of 8 waves
    ebm_phase2<<<1024, 256, 0, stream>>>(
        eps, nan_mask, e1_w, e1_b, e2_w, e2_b, e3_w, e3_b,
        mu_ws, std_ws, sls_ws, tp_ws, es_ws, liw_ws);

    // 2048 (b,s) waves = 256 blocks of 8 waves
    ebm_phase3<<<256, 256, 0, stream>>>(
        gumbel_u, eps, mu_ws, std_ws, es_ws, liw_ws, z_samples, log_probs);
}
